// CausalWanModel_62878321213727
// MI455X (gfx1250) — matrix operand order repaired and checked by reading, not compiled, not run on hardware
//
#include <hip/hip_runtime.h>
#include <cstddef>

// ---------------------------------------------------------------------------
// Problem constants (from reference)
// ---------------------------------------------------------------------------
#define DIMX   1536
#define HEADS  12
#define HD     128
#define TSEQ   1560
#define TPAD   1568          // TSEQ padded to multiple of 16
#define SKEYS  9360          // attention window length (le - ws)
#define SPAD   9376          // padded to multiple of 32
#define NEWOFF 7800          // ls: where new K/V rows land
#define SINKN  1560
#define EVICT  1560
#define QT_PER_BLK 4

typedef __bf16 bf16;
typedef __attribute__((ext_vector_type(16))) __bf16 v16bf;
typedef __attribute__((ext_vector_type(8)))  __bf16 v8bf;
typedef __attribute__((ext_vector_type(8)))  float  v8f;
typedef __attribute__((ext_vector_type(4)))  unsigned int u32x4;
typedef int v4i __attribute__((vector_size(16)));   // matches builtin param type

#define AS1 __attribute__((address_space(1)))
#define AS3 __attribute__((address_space(3)))

#if __has_builtin(__builtin_amdgcn_global_load_async_to_lds_b128)
#define HAVE_ASYNC_LDS 1
#else
#define HAVE_ASYNC_LDS 0
#endif

#if HAVE_ASYNC_LDS
// generic -> AS1 / AS3 via integer round-trip (LDS generic addr low 32 bits
// are the wave-relative LDS offset per the CDNA5 aperture rules)
__device__ __forceinline__ AS1 v4i* as_global_v4(const void* p) {
    return (AS1 v4i*)(unsigned long long)(size_t)p;
}
__device__ __forceinline__ AS3 v4i* as_lds_v4(void* p) {
    return (AS3 v4i*)(unsigned)(size_t)p;
}
#endif

// A-fragment: lane holds row m = lane&15; elems 0..15 = K[lhalf*16 .. +15]
__device__ __forceinline__ v16bf load_afrag(const bf16* p0) {
    return *(const v16bf*)p0;
}
// B-fragment: lane holds col n = lane&15; elems 0..15 = K[lhalf*16 .. +15]
__device__ __forceinline__ v16bf load_bfrag(const bf16* p) {
    return *(const v16bf*)p;
}
__device__ __forceinline__ v8f wmma_bf16(v16bf a, v16bf b, v8f c) {
    return __builtin_amdgcn_wmma_f32_16x16x32_bf16(false, a, false, b,
                                                   (short)0, c, false, false);
}

// ---------------------------------------------------------------------------
// f32 -> bf16 elementwise convert
// ---------------------------------------------------------------------------
__global__ __launch_bounds__(256) void cvt_bf16_kernel(const float* __restrict__ in,
                                                       bf16* __restrict__ out, int n) {
    int i = blockIdx.x * 256 + threadIdx.x;
    if (i < n) out[i] = (bf16)in[i];
}

// ---------------------------------------------------------------------------
// GEMM: out[m,n] = sum_k A[m,k] * W[n,k] + bias[n]      (Y = A * W^T + b)
// block = 128 threads (4 waves); wave tile 32(M) x 64(N); grid (TPAD/32, 6)
// ---------------------------------------------------------------------------
__global__ __launch_bounds__(128) void gemm_bf16_kernel(const bf16* __restrict__ A,
                                                        const bf16* __restrict__ W,
                                                        const float* __restrict__ bias,
                                                        float* __restrict__ out,
                                                        int mrows) {
    const int wave = threadIdx.x >> 5;
    const int lane = threadIdx.x & 31;
    const int lrow = lane & 15, lhalf = lane >> 4;
    const int m0 = blockIdx.x * 32;
    const int n0 = (blockIdx.y * 4 + wave) * 64;

    v8f c[2][4] = {};
    for (int k0 = 0; k0 < DIMX; k0 += 32) {
        v16bf a[2], b[4];
#pragma unroll
        for (int mi = 0; mi < 2; ++mi) {
            const bf16* p = A + (size_t)(m0 + mi * 16 + lrow) * DIMX + k0 + lhalf * 16;
            __builtin_prefetch(p + 32, 0, 1);
            a[mi] = load_afrag(p);
        }
#pragma unroll
        for (int ni = 0; ni < 4; ++ni) {
            const bf16* p = W + (size_t)(n0 + ni * 16 + lrow) * DIMX + k0 + lhalf * 16;
            __builtin_prefetch(p + 32, 0, 1);
            b[ni] = load_bfrag(p);
        }
#pragma unroll
        for (int mi = 0; mi < 2; ++mi)
#pragma unroll
            for (int ni = 0; ni < 4; ++ni)
                c[mi][ni] = wmma_bf16(a[mi], b[ni], c[mi][ni]);
    }
    // epilogue: C layout -> lane holds col n = lane&15, rows lhalf*8 + r
    if (m0 + 32 <= mrows) {           // fast path: whole tile in range
#pragma unroll
        for (int ni = 0; ni < 4; ++ni) {
            const int col = n0 + ni * 16 + lrow;
            const float bv = bias[col];
#pragma unroll
            for (int mi = 0; mi < 2; ++mi)
#pragma unroll
                for (int r = 0; r < 8; ++r) {
                    const int row = m0 + mi * 16 + lhalf * 8 + r;
                    out[(size_t)row * DIMX + col] = c[mi][ni][r] + bv;
                }
        }
    } else {
#pragma unroll
        for (int ni = 0; ni < 4; ++ni) {
            const int col = n0 + ni * 16 + lrow;
            const float bv = bias[col];
#pragma unroll
            for (int mi = 0; mi < 2; ++mi)
#pragma unroll
                for (int r = 0; r < 8; ++r) {
                    const int row = m0 + mi * 16 + lhalf * 8 + r;
                    if (row < mrows) out[(size_t)row * DIMX + col] = c[mi][ni][r] + bv;
                }
        }
    }
}

// ---------------------------------------------------------------------------
// Fused RMSNorm + RoPE. One block per token row.
// is_k == 0: write bf16 to qb[t][d];  is_k == 1: write to Kc[h][NEWOFF+t][hd]
// ---------------------------------------------------------------------------
__global__ __launch_bounds__(256) void norm_rope_kernel(const float* __restrict__ xin,
                                                        const float* __restrict__ g,
                                                        const float* __restrict__ fcos,
                                                        const float* __restrict__ fsin,
                                                        bf16* __restrict__ dst,
                                                        int is_k) {
    const int t = blockIdx.x;
    const float* row = xin + (size_t)t * DIMX;
    float ss = 0.f;
    for (int d = threadIdx.x; d < DIMX; d += 256) { float v = row[d]; ss += v * v; }
#pragma unroll
    for (int off = 16; off > 0; off >>= 1) ss += __shfl_xor(ss, off, 32);
    __shared__ float red[8];
    if ((threadIdx.x & 31) == 0) red[threadIdx.x >> 5] = ss;
    __syncthreads();
    float tot = 0.f;
#pragma unroll
    for (int w = 0; w < 8; ++w) tot += red[w];
    const float rms = rsqrtf(tot * (1.0f / DIMX) + 1e-6f);

    for (int d = threadIdx.x; d < DIMX; d += 256) {
        const int h = d >> 7, r = d & 127;
        const int i = (r < 64) ? r : r - 64;
        const float c = fcos[t * 64 + i];
        const float s = fsin[t * 64 + i];
        const int b = h * HD;
        const float t1 = row[b + i]      * rms * g[b + i];
        const float t2 = row[b + 64 + i] * rms * g[b + 64 + i];
        const float val = (r < 64) ? (t1 * c - t2 * s) : (t1 * s + t2 * c);
        if (is_k)
            dst[((size_t)h * SPAD + (NEWOFF + t)) * HD + r] = (bf16)val;
        else
            dst[(size_t)t * DIMX + d] = (bf16)val;
    }
}

// ---------------------------------------------------------------------------
// V (f32, post-bias) -> Vt bf16 [h][hd][NEWOFF+t]
// ---------------------------------------------------------------------------
__global__ __launch_bounds__(256) void scatter_v_kernel(const float* __restrict__ v32,
                                                        bf16* __restrict__ Vt) {
    int idx = blockIdx.x * 256 + threadIdx.x;
    if (idx >= TSEQ * DIMX) return;
    const int t = idx / DIMX, d = idx % DIMX;
    const int h = d >> 7, r = d & 127;
    Vt[((size_t)h * HD + r) * SPAD + (NEWOFF + t)] = (bf16)v32[idx];
}

// ---------------------------------------------------------------------------
// Cache roll + convert: K -> [h][s][hd], V -> [h][hd][s]
// ---------------------------------------------------------------------------
__global__ __launch_bounds__(256) void build_cache_kernel(const float* __restrict__ ck,
                                                          const float* __restrict__ cv,
                                                          bf16* __restrict__ Kc,
                                                          bf16* __restrict__ Vt) {
    int idx = blockIdx.x * 256 + threadIdx.x;
    if (idx >= NEWOFF * DIMX) return;
    const int s = idx / DIMX, d = idx % DIMX;
    const int src = (s < SINKN) ? s : s + EVICT;
    const int h = d >> 7, r = d & 127;
    const size_t g = (size_t)src * DIMX + d;
    Kc[((size_t)h * SPAD + s) * HD + r] = (bf16)ck[g];
    Vt[((size_t)h * HD + r) * SPAD + s] = (bf16)cv[g];
}

// ---------------------------------------------------------------------------
// Flash attention. Block = 4 waves = 4 query tiles of 16, one head.
// Each 32-key chunk of K (8KB) and V (8KB) is staged in LDS cooperatively via
// async global->LDS loads (ASYNCcnt path), then consumed by WMMA from LDS.
// grid (ceil(98/4)=25, 12), block 128.
// ---------------------------------------------------------------------------
__global__ __launch_bounds__(128) void attn_kernel(const bf16* __restrict__ qb,
                                                   const bf16* __restrict__ Kc,
                                                   const bf16* __restrict__ Vt,
                                                   bf16* __restrict__ ab) {
    __shared__ __align__(32) bf16 Kl[32 * HD];            // [s_local][hd]
    __shared__ __align__(32) bf16 Vl[HD * 32];            // [hd][s_local]
    __shared__ __align__(32) bf16 Pl[QT_PER_BLK][16 * 32];

    const int h    = blockIdx.y;
    const int wave = threadIdx.x >> 5;
    const int lane = threadIdx.x & 31;
    const int tid  = threadIdx.x;
    const int lrow = lane & 15, lhalf = lane >> 4;
    int qt = blockIdx.x * QT_PER_BLK + wave;
    qt = (qt > TPAD / 16 - 1) ? (TPAD / 16 - 1) : qt;     // clamp: dup work, same data
    const float scale = 0.08838834764831845f;             // 1/sqrt(128)

    // Q A-fragments: 16 queries x 128 hd, 4 K-chunks of 32
    v16bf aq[4];
#pragma unroll
    for (int kc = 0; kc < 4; ++kc) {
        const bf16* p = qb + (size_t)(qt * 16 + lrow) * DIMX + h * HD + kc * 32 + lhalf * 16;
        aq[kc] = load_afrag(p);
    }

    float m[8], lsum[8];
#pragma unroll
    for (int r = 0; r < 8; ++r) { m[r] = -3.0e38f; lsum[r] = 0.f; }
    v8f o[8] = {};

    const char* KhB = (const char*)(Kc + (size_t)h * SPAD * HD);  // rows of 256B
    const char* VhB = (const char*)(Vt + (size_t)h * HD * SPAD);  // rows of 2*SPAD B
    bf16* myP = &Pl[wave][0];

    for (int s0 = 0; s0 < SKEYS; s0 += 32) {
        __syncthreads();   // previous chunk fully consumed by all waves
        // ---- stage K chunk (contiguous 8KB) and V chunk (128 rows x 64B)
#if HAVE_ASYNC_LDS
#pragma unroll
        for (int i = 0; i < 4; ++i) {
            const int idx = i * 128 + tid;                 // 0..511 x 16B
            __builtin_amdgcn_global_load_async_to_lds_b128(
                as_global_v4(KhB + (size_t)s0 * 256 + (size_t)idx * 16),
                as_lds_v4((char*)Kl + idx * 16), 0, 0);
            const int hd = idx >> 2, part = idx & 3;
            __builtin_amdgcn_global_load_async_to_lds_b128(
                as_global_v4(VhB + (size_t)hd * (SPAD * 2) + (size_t)s0 * 2 + part * 16),
                as_lds_v4((char*)Vl + idx * 16), 0, 0);
        }
        asm volatile("s_wait_asynccnt 0x0" ::: "memory");
#else
#pragma unroll
        for (int i = 0; i < 4; ++i) {
            const int idx = i * 128 + tid;
            *(u32x4*)((char*)Kl + idx * 16) =
                *(const u32x4*)(KhB + (size_t)s0 * 256 + (size_t)idx * 16);
            const int hd = idx >> 2, part = idx & 3;
            *(u32x4*)((char*)Vl + idx * 16) =
                *(const u32x4*)(VhB + (size_t)hd * (SPAD * 2) + (size_t)s0 * 2 + part * 16);
        }
#endif
        __syncthreads();   // chunk visible to all waves

        // ---- scores: 16 x 32 tile = two 16x16 C frags, 4 wmma each over hd
        v8f sc[2] = {};
#pragma unroll
        for (int j = 0; j < 2; ++j) {
            const bf16* kp = Kl + (j * 16 + lrow) * HD + lhalf * 16;
#pragma unroll
            for (int kc = 0; kc < 4; ++kc)
                sc[j] = wmma_bf16(aq[kc], load_bfrag(kp + kc * 32), sc[j]);
        }
        // scale + mask tail columns (only j=1 of the last chunk can exceed)
        const bool mask1 = (s0 + 16 + lrow) >= SKEYS;
#pragma unroll
        for (int r = 0; r < 8; ++r) {
            sc[0][r] *= scale;
            sc[1][r] = mask1 ? -3.0e38f : sc[1][r] * scale;
        }
        // ---- online softmax (rows replicated across each 16-lane group)
        float cm[8];
#pragma unroll
        for (int r = 0; r < 8; ++r) cm[r] = fmaxf(sc[0][r], sc[1][r]);
#pragma unroll
        for (int off = 1; off < 16; off <<= 1)
#pragma unroll
            for (int r = 0; r < 8; ++r) cm[r] = fmaxf(cm[r], __shfl_xor(cm[r], off, 32));
        float alpha[8];
#pragma unroll
        for (int r = 0; r < 8; ++r) {
            const float mn = fmaxf(m[r], cm[r]);
            alpha[r] = __expf(m[r] - mn);
            m[r] = mn;
        }
        float p0[8], p1[8], rs[8];
#pragma unroll
        for (int r = 0; r < 8; ++r) {
            p0[r] = __expf(sc[0][r] - m[r]);
            p1[r] = __expf(sc[1][r] - m[r]);
            rs[r] = p0[r] + p1[r];
        }
#pragma unroll
        for (int off = 1; off < 16; off <<= 1)
#pragma unroll
            for (int r = 0; r < 8; ++r) rs[r] += __shfl_xor(rs[r], off, 32);
#pragma unroll
        for (int r = 0; r < 8; ++r) lsum[r] = lsum[r] * alpha[r] + rs[r];
#pragma unroll
        for (int f = 0; f < 8; ++f)
#pragma unroll
            for (int r = 0; r < 8; ++r) o[f][r] *= alpha[r];

        // ---- P: C layout -> per-wave LDS -> A layout
#pragma unroll
        for (int r = 0; r < 8; ++r) {
            myP[(lhalf * 8 + r) * 32 + lrow]      = (bf16)p0[r];
            myP[(lhalf * 8 + r) * 32 + 16 + lrow] = (bf16)p1[r];
        }
        asm volatile("s_wait_dscnt 0x0" ::: "memory");
        const v16bf pa = load_afrag(&myP[lrow * 32 + lhalf * 16]);
        asm volatile("" ::: "memory");

        // ---- P * V : 8 output hd-tiles of 16
#pragma unroll
        for (int f = 0; f < 8; ++f) {
            const bf16* vp = Vl + (f * 16 + lrow) * 32 + lhalf * 16;
            o[f] = wmma_bf16(pa, load_bfrag(vp), o[f]);
        }
    }

    // ---- finalize + store bf16 attention output [t][h*HD + hd]
    float inv[8];
#pragma unroll
    for (int r = 0; r < 8; ++r) inv[r] = 1.0f / lsum[r];
#pragma unroll
    for (int f = 0; f < 8; ++f) {
#pragma unroll
        for (int r = 0; r < 8; ++r) {
            const int t = qt * 16 + lhalf * 8 + r;
            if (t < TSEQ)
                ab[(size_t)t * DIMX + h * HD + f * 16 + lrow] = (bf16)(o[f][r] * inv[r]);
        }
    }
}

// ---------------------------------------------------------------------------
// Host launcher
// ---------------------------------------------------------------------------
extern "C" void kernel_launch(void* const* d_in, const int* in_sizes, int n_in,
                              void* d_out, int out_size, void* d_ws, size_t ws_size,
                              hipStream_t stream) {
    (void)in_sizes; (void)n_in; (void)out_size; (void)ws_size;
    const float* x  = (const float*)d_in[0];
    const float* fc = (const float*)d_in[1];
    const float* fs = (const float*)d_in[2];
    const float* ck = (const float*)d_in[3];
    const float* cv = (const float*)d_in[4];
    const float* wq = (const float*)d_in[5];
    const float* bq = (const float*)d_in[6];
    const float* wk = (const float*)d_in[7];
    const float* bk = (const float*)d_in[8];
    const float* wv = (const float*)d_in[9];
    const float* bv = (const float*)d_in[10];
    const float* wo = (const float*)d_in[11];
    const float* bo = (const float*)d_in[12];
    const float* gq = (const float*)d_in[13];
    const float* gk = (const float*)d_in[14];
    float* out = (float*)d_out;
    char*  ws  = (char*)d_ws;

    const size_t SZ_XB  = (size_t)TPAD * DIMX * 2;
    const size_t SZ_W   = (size_t)DIMX * DIMX * 2;
    const size_t SZ_F32 = (size_t)TPAD * DIMX * 4;
    const size_t SZ_KV  = (size_t)HEADS * SPAD * HD * 2;
    size_t off = 0;
    bf16* xb  = (bf16*)(ws + off); off += SZ_XB;
    bf16* wqb = (bf16*)(ws + off); off += SZ_W;
    bf16* wkb = (bf16*)(ws + off); off += SZ_W;
    bf16* wvb = (bf16*)(ws + off); off += SZ_W;
    bf16* wob = (bf16*)(ws + off); off += SZ_W;
    float* q32 = (float*)(ws + off); off += SZ_F32;
    float* k32 = (float*)(ws + off); off += SZ_F32;
    float* v32 = (float*)(ws + off); off += SZ_F32;
    bf16* qb  = (bf16*)(ws + off); off += SZ_XB;
    bf16* Kc  = (bf16*)(ws + off); off += SZ_KV;
    bf16* Vt  = (bf16*)(ws + off); off += SZ_KV;
    bf16* ab  = (bf16*)(ws + off); off += SZ_XB;

    const int nX = TSEQ * DIMX;
    const int nW = DIMX * DIMX;
    cvt_bf16_kernel<<<(nX + 255) / 256, 256, 0, stream>>>(x,  xb,  nX);
    cvt_bf16_kernel<<<(nW + 255) / 256, 256, 0, stream>>>(wq, wqb, nW);
    cvt_bf16_kernel<<<(nW + 255) / 256, 256, 0, stream>>>(wk, wkb, nW);
    cvt_bf16_kernel<<<(nW + 255) / 256, 256, 0, stream>>>(wv, wvb, nW);
    cvt_bf16_kernel<<<(nW + 255) / 256, 256, 0, stream>>>(wo, wob, nW);

    dim3 ggrid(TPAD / 32, DIMX / 256), gblk(128);
    gemm_bf16_kernel<<<ggrid, gblk, 0, stream>>>(xb, wqb, bq, q32, TPAD);
    gemm_bf16_kernel<<<ggrid, gblk, 0, stream>>>(xb, wkb, bk, k32, TPAD);
    gemm_bf16_kernel<<<ggrid, gblk, 0, stream>>>(xb, wvb, bv, v32, TPAD);

    norm_rope_kernel<<<TSEQ, 256, 0, stream>>>(q32, gq, fc, fs, qb, 0);
    norm_rope_kernel<<<TSEQ, 256, 0, stream>>>(k32, gk, fc, fs, Kc, 1);
    scatter_v_kernel<<<(TSEQ * DIMX + 255) / 256, 256, 0, stream>>>(v32, Vt);

    build_cache_kernel<<<(NEWOFF * DIMX + 255) / 256, 256, 0, stream>>>(ck, cv, Kc, Vt);

    const int nqt = TPAD / 16;                               // 98
    attn_kernel<<<dim3((nqt + QT_PER_BLK - 1) / QT_PER_BLK, HEADS), 128, 0, stream>>>(
        qb, Kc, Vt, ab);

    gemm_bf16_kernel<<<ggrid, gblk, 0, stream>>>(ab, wob, bo, out, TSEQ);
}
